// VGNN_sparse_59571196396043
// MI455X (gfx1250) — matrix operand order repaired and checked by reading, not compiled, not run on hardware
//
#include <hip/hip_runtime.h>
#include <hip/hip_bf16.h>
#include <math.h>

// ---------------- problem constants ----------------
#define N_NODES   10000
#define N_FACTORS 20000
#define SEGS      (N_NODES + N_FACTORS)   // 30000
#define N_EDGES   (4 * N_FACTORS)         // 80000
#define STATE_DIM 64
#define MSG_DIM   64
#define HID_MSG   128
#define HID_RO    128
#define D_IN_PAD  160                     // 2*64+4 = 132 padded to multiple of 32
#define N_STEPS   10
#define ETILE     64                      // edges per block

typedef __attribute__((ext_vector_type(16))) _Float16 v16h;
typedef __attribute__((ext_vector_type(8)))  _Float16 v8h;
typedef __attribute__((ext_vector_type(8)))  float    v8f;
typedef __attribute__((ext_vector_type(4)))  int      v4i;

#define WMMA_F16(a, b, c) \
  __builtin_amdgcn_wmma_f32_16x16x32_f16(false, (a), false, (b), (short)0, (c), false, false)

// ---------------- CDNA5 async global->LDS copy (guarded) ----------------
#ifndef __has_builtin
#define __has_builtin(x) 0
#endif

#if __has_builtin(__builtin_amdgcn_global_load_async_to_lds_b128)
#define HAVE_ASYNC_LDS 1
#else
#define HAVE_ASYNC_LDS 0
#endif

// builtin expects pointers to 16-byte int vectors: global src, LDS dst
typedef __attribute__((address_space(1))) v4i v4i_g;
typedef __attribute__((address_space(3))) v4i v4i_l;

// copy 16 bytes (8 halves) global -> LDS, per-lane addresses
__device__ __forceinline__ void async_copy16(const _Float16* g, _Float16* l) {
#if HAVE_ASYNC_LDS
  __builtin_amdgcn_global_load_async_to_lds_b128((v4i_g*)g, (v4i_l*)l, 0, 0);
#else
  *(v8h*)l = *(const v8h*)g;
#endif
}

__device__ __forceinline__ void async_wait0() {
#if HAVE_ASYNC_LDS
#if __has_builtin(__builtin_amdgcn_s_wait_asynccnt)
  __builtin_amdgcn_s_wait_asynccnt(0);
#else
  asm volatile("s_wait_asynccnt 0x0" ::: "memory");
#endif
#endif
}

// ---------------- WMMA fragment loaders ----------------
// A (16x32, f16). ISA layout: lanes 0-15 -> M=lane, K={0..7, 16..23};
// lanes 16-31 -> M=lane-16, K={8..15, 24..31}.
__device__ __forceinline__ v16h load_frag_a(const _Float16* base, int ld) {
  int lane = threadIdx.x & 31;
  int m    = lane & 15;
  int koff = (lane & 16) ? 8 : 0;
  const _Float16* p = base + m * ld + koff;
  v8h lo = *(const v8h*)(p);
  v8h hi = *(const v8h*)(p + 16);
  return __builtin_shufflevector(lo, hi, 0,1,2,3,4,5,6,7,8,9,10,11,12,13,14,15);
}

// B (32x16, f16), B[k][n] = W[n0+n][k0+k] for row-major W (out x in).
// lanes 0-15 hold K={0..7, 16..23} for N=lane; lanes 16-31 hold K={8..15, 24..31} for N=lane-16.
__device__ __forceinline__ v16h load_frag_b(const _Float16* base, int ld) {
  int lane = threadIdx.x & 31;
  int n    = lane & 15;
  int koff = (lane & 16) ? 8 : 0;
  const _Float16* p = base + n * ld + koff;
  v8h lo = *(const v8h*)(p);
  v8h hi = *(const v8h*)(p + 16);
  return __builtin_shufflevector(lo, hi, 0,1,2,3,4,5,6,7,8,9,10,11,12,13,14,15);
}

__device__ __forceinline__ float sigmoidf_(float x) { return 1.0f / (1.0f + __expf(-x)); }

// ---------------- small utility kernels ----------------
__global__ void cvt_pad_f16(const float* __restrict__ src, _Float16* __restrict__ dst,
                            int rows, int cs, int cd) {
  int idx = blockIdx.x * blockDim.x + threadIdx.x;
  if (idx >= rows * cd) return;
  int r = idx / cd, c = idx - r * cd;
  dst[idx] = (c < cs) ? (_Float16)src[r * cs + c] : (_Float16)0.0f;
}

__global__ void init_state_kernel(_Float16* __restrict__ h16, float* __restrict__ c) {
  int idx = blockIdx.x * blockDim.x + threadIdx.x;
  if (idx >= SEGS * STATE_DIM) return;
  int seg = idx >> 6, d = idx & 63;
  c[idx]   = 0.0f;
  h16[idx] = (seg >= N_NODES && d == 0) ? (_Float16)1.0f : (_Float16)0.0f;
}

__global__ void zero_f32_kernel(float* __restrict__ p, int n) {
  int idx = blockIdx.x * blockDim.x + threadIdx.x;
  if (idx < n) p[idx] = 0.0f;
}

// ---------------- edge MLP + segment-sum scatter ----------------
// 64 edges / block, 256 threads = 8 waves. Wave w owns N-tile w; weight (B)
// fragments loaded once per K-slice and reused across 4 M-tiles.
__global__ __launch_bounds__(256) void edge_kernel(
    const int* __restrict__ rowv, const int* __restrict__ colv,
    const _Float16* __restrict__ ea16, const _Float16* __restrict__ h16,
    const _Float16* __restrict__ W1, const float* __restrict__ B1,
    const _Float16* __restrict__ W2, const float* __restrict__ B2,
    const _Float16* __restrict__ W3, const float* __restrict__ B3,
    float* __restrict__ nm)
{
  __shared__ _Float16 X0[ETILE * D_IN_PAD];   // 20.0 KB gathered input (padded)
  __shared__ _Float16 X1[ETILE * HID_MSG];    // 16 KB
  __shared__ _Float16 X2[ETILE * HID_MSG];    // 16 KB
  __shared__ int scol[ETILE];

  int tid = threadIdx.x;
  int e0  = blockIdx.x * ETILE;

  // warm L0/L2 with layer-2/3 weights while the gather is in flight
  for (int p = tid; p < 384; p += 256) {
    const _Float16* a = (p < 256) ? (W2 + p * 64) : (W3 + (p - 256) * 64);
    __builtin_prefetch((const void*)a, 0, 0);
  }

  // ---- gather: 64 edges x 20 chunks of 8 halves (async global->LDS) ----
  for (int ch = tid; ch < ETILE * 20; ch += 256) {
    int e = ch / 20, cc = ch - e * 20;
    int ge = e0 + e;
    _Float16* dst = &X0[e * D_IN_PAD + cc * 8];
    if (cc < 8) {
      async_copy16(h16 + (size_t)rowv[ge] * STATE_DIM + cc * 8, dst);
    } else if (cc < 16) {
      async_copy16(h16 + (size_t)colv[ge] * STATE_DIM + (cc - 8) * 8, dst);
    } else if (cc == 16) {
      const _Float16* ea = ea16 + (size_t)ge * 4;
      v8h val = {};
      val[0] = ea[0]; val[1] = ea[1]; val[2] = ea[2]; val[3] = ea[3];
      *(v8h*)dst = val;
    } else {
      v8h z = {};
      *(v8h*)dst = z;
    }
  }
  if (tid < ETILE) scol[tid] = colv[e0 + tid];
  async_wait0();
  __syncthreads();

  int wave = tid >> 5;
  int lane = tid & 31;
  int nlo  = lane & 15;
  int mhi  = (lane & 16) ? 8 : 0;

  // ---- layer 1: (64x160) @ (160x128) -> relu -> X1. nt = wave ----
  {
    int nt = wave;
    v8f acc[4] = {};
    #pragma unroll
    for (int ks = 0; ks < D_IN_PAD / 32; ++ks) {
      v16h b = load_frag_b(W1 + (size_t)(nt * 16) * D_IN_PAD + ks * 32, D_IN_PAD);
      #pragma unroll
      for (int mt = 0; mt < 4; ++mt) {
        v16h a = load_frag_a(&X0[mt * 16 * D_IN_PAD + ks * 32], D_IN_PAD);
        acc[mt] = WMMA_F16(a, b, acc[mt]);
      }
    }
    int n = nt * 16 + nlo;
    float bias = B1[n];
    #pragma unroll
    for (int mt = 0; mt < 4; ++mt)
      #pragma unroll
      for (int r = 0; r < 8; ++r) {
        float v = acc[mt][r] + bias;
        X1[(mt * 16 + mhi + r) * HID_MSG + n] = (_Float16)(v > 0.0f ? v : 0.0f);
      }
  }
  __syncthreads();

  // ---- layer 2: (64x128) @ (128x128) -> relu -> X2. nt = wave ----
  {
    int nt = wave;
    v8f acc[4] = {};
    #pragma unroll
    for (int ks = 0; ks < HID_MSG / 32; ++ks) {
      v16h b = load_frag_b(W2 + (size_t)(nt * 16) * HID_MSG + ks * 32, HID_MSG);
      #pragma unroll
      for (int mt = 0; mt < 4; ++mt) {
        v16h a = load_frag_a(&X1[mt * 16 * HID_MSG + ks * 32], HID_MSG);
        acc[mt] = WMMA_F16(a, b, acc[mt]);
      }
    }
    int n = nt * 16 + nlo;
    float bias = B2[n];
    #pragma unroll
    for (int mt = 0; mt < 4; ++mt)
      #pragma unroll
      for (int r = 0; r < 8; ++r) {
        float v = acc[mt][r] + bias;
        X2[(mt * 16 + mhi + r) * HID_MSG + n] = (_Float16)(v > 0.0f ? v : 0.0f);
      }
  }
  __syncthreads();

  // ---- layer 3: (64x128) @ (128x64) -> scatter-add into nm by col ----
  {
    int nt  = wave & 3;
    int mtb = (wave >> 2) * 2;         // waves 0-3: M-tiles 0,1; waves 4-7: 2,3
    v8f acc[2] = {};
    #pragma unroll
    for (int ks = 0; ks < HID_MSG / 32; ++ks) {
      v16h b = load_frag_b(W3 + (size_t)(nt * 16) * HID_MSG + ks * 32, HID_MSG);
      #pragma unroll
      for (int j = 0; j < 2; ++j) {
        v16h a = load_frag_a(&X2[(mtb + j) * 16 * HID_MSG + ks * 32], HID_MSG);
        acc[j] = WMMA_F16(a, b, acc[j]);
      }
    }
    int n = nt * 16 + nlo;
    float bias = B3[n];
    #pragma unroll
    for (int j = 0; j < 2; ++j)
      #pragma unroll
      for (int r = 0; r < 8; ++r) {
        int e = (mtb + j) * 16 + mhi + r;
        atomicAdd(&nm[(size_t)scol[e] * MSG_DIM + n], acc[j][r] + bias);
      }
  }
}

// ---------------- LSTM cell over all segments ----------------
// 16 segments / block, 256 threads = 8 waves.
__global__ __launch_bounds__(256) void lstm_kernel(
    const float* __restrict__ nm, _Float16* __restrict__ h16,
    const _Float16* __restrict__ Wih, const _Float16* __restrict__ Whh,
    const float* __restrict__ bih, const float* __restrict__ bhh,
    float* __restrict__ c)
{
  __shared__ _Float16 Anm[16 * STATE_DIM];
  __shared__ _Float16 Ah [16 * STATE_DIM];
  __shared__ float    G  [16 * 4 * STATE_DIM];   // 16 x 256 gates

  int tid = threadIdx.x;
  int r0  = blockIdx.x * 16;

  // h tile is contiguous (rows r0..r0+15): async copy 16B chunks into LDS
  for (int ch = tid; ch < (16 * STATE_DIM) / 8; ch += 256)
    async_copy16(h16 + (size_t)r0 * STATE_DIM + ch * 8, &Ah[ch * 8]);
  // nm needs f32->f16 conversion: regular path
  for (int idx = tid; idx < 16 * STATE_DIM; idx += 256) {
    int m = idx >> 6, d = idx & 63;
    Anm[idx] = (_Float16)nm[(size_t)(r0 + m) * STATE_DIM + d];
  }
  async_wait0();
  __syncthreads();

  int wave = tid >> 5;
  int lane = tid & 31;
  int nlo  = lane & 15;
  int mhi  = (lane & 16) ? 8 : 0;

  for (int i = 0; i < 2; ++i) {
    int nt = wave * 2 + i;   // 16 N-tiles over 256 gate outputs
    v8f acc = {};
    #pragma unroll
    for (int ks = 0; ks < STATE_DIM / 32; ++ks) {
      v16h a1 = load_frag_a(&Anm[ks * 32], STATE_DIM);
      v16h b1 = load_frag_b(Wih + (size_t)(nt * 16) * STATE_DIM + ks * 32, STATE_DIM);
      acc = WMMA_F16(a1, b1, acc);
      v16h a2 = load_frag_a(&Ah[ks * 32], STATE_DIM);
      v16h b2 = load_frag_b(Whh + (size_t)(nt * 16) * STATE_DIM + ks * 32, STATE_DIM);
      acc = WMMA_F16(a2, b2, acc);
    }
    int n = nt * 16 + nlo;
    float bias = bih[n] + bhh[n];
    #pragma unroll
    for (int r = 0; r < 8; ++r) G[(mhi + r) * 256 + n] = acc[r] + bias;
  }
  __syncthreads();

  for (int idx = tid; idx < 16 * STATE_DIM; idx += 256) {
    int m = idx >> 6, d = idx & 63;
    float gi = G[m * 256 + d];
    float gf = G[m * 256 + 64 + d];
    float gg = G[m * 256 + 128 + d];
    float go = G[m * 256 + 192 + d];
    size_t off = (size_t)(r0 + m) * STATE_DIM + d;
    float cn = sigmoidf_(gf) * c[off] + sigmoidf_(gi) * tanhf(gg);
    float hn = sigmoidf_(go) * tanhf(cn);
    c[off]   = cn;
    h16[off] = (_Float16)hn;
  }
}

// ---------------- readout MLP + softmax ----------------
// 16 nodes / block, 256 threads = 8 waves.
__global__ __launch_bounds__(256) void readout_kernel(
    const _Float16* __restrict__ h16,
    const _Float16* __restrict__ RW1, const float* __restrict__ RB1,
    const _Float16* __restrict__ RW2, const float* __restrict__ RB2,
    const float* __restrict__ RW3, const float* __restrict__ RB3,
    float* __restrict__ out)
{
  __shared__ _Float16 A0[16 * STATE_DIM];
  __shared__ _Float16 X1[16 * HID_RO];
  __shared__ _Float16 X2[16 * HID_RO];

  int tid = threadIdx.x;
  int n0  = blockIdx.x * 16;

  for (int ch = tid; ch < (16 * STATE_DIM) / 8; ch += 256)
    async_copy16(h16 + (size_t)n0 * STATE_DIM + ch * 8, &A0[ch * 8]);
  async_wait0();
  __syncthreads();

  int wave = tid >> 5;
  int lane = tid & 31;
  int nlo  = lane & 15;
  int mhi  = (lane & 16) ? 8 : 0;

  // layer 1: (16x64) @ (64x128) -> relu
  {
    int nt = wave;
    v8f acc = {};
    #pragma unroll
    for (int ks = 0; ks < STATE_DIM / 32; ++ks) {
      v16h a = load_frag_a(&A0[ks * 32], STATE_DIM);
      v16h b = load_frag_b(RW1 + (size_t)(nt * 16) * STATE_DIM + ks * 32, STATE_DIM);
      acc = WMMA_F16(a, b, acc);
    }
    int n = nt * 16 + nlo;
    float bias = RB1[n];
    #pragma unroll
    for (int r = 0; r < 8; ++r) {
      float v = acc[r] + bias;
      X1[(mhi + r) * HID_RO + n] = (_Float16)(v > 0.0f ? v : 0.0f);
    }
  }
  __syncthreads();

  // layer 2: (16x128) @ (128x128) -> relu
  {
    int nt = wave;
    v8f acc = {};
    #pragma unroll
    for (int ks = 0; ks < HID_RO / 32; ++ks) {
      v16h a = load_frag_a(&X1[ks * 32], HID_RO);
      v16h b = load_frag_b(RW2 + (size_t)(nt * 16) * HID_RO + ks * 32, HID_RO);
      acc = WMMA_F16(a, b, acc);
    }
    int n = nt * 16 + nlo;
    float bias = RB2[n];
    #pragma unroll
    for (int r = 0; r < 8; ++r) {
      float v = acc[r] + bias;
      X2[(mhi + r) * HID_RO + n] = (_Float16)(v > 0.0f ? v : 0.0f);
    }
  }
  __syncthreads();

  // final 2-logit head + softmax (tiny: scalar per node)
  if (tid < 16) {
    int node = n0 + tid;
    if (node < N_NODES) {
      float z0 = RB3[0], z1 = RB3[1];
      for (int k = 0; k < HID_RO; ++k) {
        float x = (float)X2[tid * HID_RO + k];
        z0 += x * RW3[k];
        z1 += x * RW3[HID_RO + k];
      }
      float mx = fmaxf(z0, z1);
      float ex0 = __expf(z0 - mx), ex1 = __expf(z1 - mx);
      float inv = 1.0f / (ex0 + ex1);
      out[(size_t)node * 2]     = ex0 * inv;
      out[(size_t)node * 2 + 1] = ex1 * inv;
    }
  }
}

// ---------------- launcher ----------------
extern "C" void kernel_launch(void* const* d_in, const int* in_sizes, int n_in,
                              void* d_out, int out_size, void* d_ws, size_t ws_size,
                              hipStream_t stream) {
  const int*   rowv      = (const int*)d_in[0];
  const int*   colv      = (const int*)d_in[1];
  const float* edge_attr = (const float*)d_in[2];
  // d_in[3] = n_nodes, d_in[4] = num_segments (compile-time constants here)
  const float* mp_w1 = (const float*)d_in[5];
  const float* mp_b1 = (const float*)d_in[6];
  const float* mp_w2 = (const float*)d_in[7];
  const float* mp_b2 = (const float*)d_in[8];
  const float* mp_w3 = (const float*)d_in[9];
  const float* mp_b3 = (const float*)d_in[10];
  const float* w_ih  = (const float*)d_in[11];
  const float* w_hh  = (const float*)d_in[12];
  const float* b_ih  = (const float*)d_in[13];
  const float* b_hh  = (const float*)d_in[14];
  const float* ro_w1 = (const float*)d_in[15];
  const float* ro_b1 = (const float*)d_in[16];
  const float* ro_w2 = (const float*)d_in[17];
  const float* ro_b2 = (const float*)d_in[18];
  const float* ro_w3 = (const float*)d_in[19];
  const float* ro_b3 = (const float*)d_in[20];
  (void)in_sizes; (void)n_in; (void)out_size; (void)ws_size;

  char* ws = (char*)d_ws;
  size_t off = 0;
  auto alloc = [&](size_t bytes) -> void* {
    void* p = ws + off;
    off = (off + bytes + 255) & ~(size_t)255;
    return p;
  };

  _Float16* h16  = (_Float16*)alloc((size_t)SEGS * STATE_DIM * 2);
  float*    cst  = (float*)   alloc((size_t)SEGS * STATE_DIM * 4);
  float*    nm   = (float*)   alloc((size_t)SEGS * MSG_DIM * 4);
  _Float16* ea16 = (_Float16*)alloc((size_t)N_EDGES * 4 * 2);
  _Float16* W1   = (_Float16*)alloc((size_t)HID_MSG * D_IN_PAD * 2);
  _Float16* W2   = (_Float16*)alloc((size_t)HID_MSG * HID_MSG * 2);
  _Float16* W3   = (_Float16*)alloc((size_t)MSG_DIM * HID_MSG * 2);
  _Float16* Wih  = (_Float16*)alloc((size_t)4 * STATE_DIM * MSG_DIM * 2);
  _Float16* Whh  = (_Float16*)alloc((size_t)4 * STATE_DIM * STATE_DIM * 2);
  _Float16* RW1  = (_Float16*)alloc((size_t)HID_RO * STATE_DIM * 2);
  _Float16* RW2  = (_Float16*)alloc((size_t)HID_RO * HID_RO * 2);

  auto cvt = [&](const float* s, _Float16* d, int r, int cs, int cd) {
    int n = r * cd;
    cvt_pad_f16<<<(n + 255) / 256, 256, 0, stream>>>(s, d, r, cs, cd);
  };
  cvt(mp_w1, W1, HID_MSG, 2 * STATE_DIM + 4, D_IN_PAD);
  cvt(mp_w2, W2, HID_MSG, HID_MSG, HID_MSG);
  cvt(mp_w3, W3, MSG_DIM, HID_MSG, HID_MSG);
  cvt(w_ih,  Wih, 4 * STATE_DIM, MSG_DIM, MSG_DIM);
  cvt(w_hh,  Whh, 4 * STATE_DIM, STATE_DIM, STATE_DIM);
  cvt(ro_w1, RW1, HID_RO, STATE_DIM, STATE_DIM);
  cvt(ro_w2, RW2, HID_RO, HID_RO, HID_RO);
  cvt(edge_attr, ea16, N_EDGES, 4, 4);

  {
    int n = SEGS * STATE_DIM;
    init_state_kernel<<<(n + 255) / 256, 256, 0, stream>>>(h16, cst);
  }

  for (int step = 0; step < N_STEPS; ++step) {
    int n = SEGS * MSG_DIM;
    zero_f32_kernel<<<(n + 255) / 256, 256, 0, stream>>>(nm, n);
    edge_kernel<<<N_EDGES / ETILE, 256, 0, stream>>>(rowv, colv, ea16, h16,
                                                     W1, mp_b1, W2, mp_b2, W3, mp_b3, nm);
    lstm_kernel<<<SEGS / 16, 256, 0, stream>>>(nm, h16, Wih, Whh, b_ih, b_hh, cst);
  }

  readout_kernel<<<(N_NODES + 15) / 16, 256, 0, stream>>>(
      h16, RW1, ro_b1, RW2, ro_b2, ro_w3, ro_b3, (float*)d_out);
}
